// Reformer_Combiner_74629351735746
// MI455X (gfx1250) — matrix operand order repaired and checked by reading, not compiled, not run on hardware
//
#include <hip/hip_runtime.h>
#include <math.h>

typedef __attribute__((ext_vector_type(16))) _Float16 v16h;
typedef __attribute__((ext_vector_type(8)))  _Float16 v8h;
typedef __attribute__((ext_vector_type(4)))  _Float16 v4h;
typedef __attribute__((ext_vector_type(8)))  float    v8f;

// ---------------- model constants ----------------
static constexpr int cB = 4, cT = 2048, cIN = 1024, cDIM = 512;
static constexpr int cLH = 2, cSH = 6, cNH = 4, cBUCKET = 64;
static constexpr int cDH = 64;                 // DIM / HEADS
static constexpr int cNB = cT / cBUCKET;       // 32 buckets per hash
static constexpr int cNCH = cNH * cNB;         // 128 chunks per lsh row
static constexpr int cROWS = cB * cT;          // 8192 token rows
static constexpr int cBhL = cB * cSH;          // 24 lsh batch-head rows
static constexpr int cNJ = cNH * cT;           // 8192 sort elements per row
static constexpr int cWIN = 2 * cBUCKET;       // 128 local window
static constexpr int cNW = cT / cWIN;          // 16 windows
static constexpr float MASKV = -1e9f, SELFV = -5e4f;

// ---------------- workspace layout (float offsets) ----------------
static constexpr size_t F_X    = (size_t)cROWS * cDIM;           // 4,194,304
static constexpr size_t OFF_X1 = 0;
static constexpr size_t OFF_X2 = F_X;
static constexpr size_t OFF_H  = 2 * F_X;
static constexpr size_t OFF_QK = 3 * F_X;
static constexpr size_t OFF_V  = 4 * F_X;
static constexpr size_t OFF_ATT= 5 * F_X;
static constexpr size_t OFF_BIG= 6 * F_X;                        // FF hidden / LSH O union
static constexpr size_t SZ_BIG = (size_t)cROWS * 4 * cDIM;       // 16,777,216
static constexpr size_t OFF_O  = OFF_BIG;                        // 12,582,912 used
static constexpr size_t OFF_LSE= OFF_BIG + (size_t)cBhL * cNH * cT * cDH;
static constexpr size_t OFF_ROT= OFF_BIG + SZ_BIG;
static constexpr size_t OFF_SK = OFF_ROT + 4096;
static constexpr size_t OFF_ST = OFF_SK + (size_t)cBhL * cNJ;

// =====================================================================
// WMMA GEMM: C[M,N] = act(A[M,K] @ W[K,N] + bias) [optionally C += ...]
// block 256 thr = 8 waves (2x4), tile 32x64, Kstep 64 (2 wmma / barrier).
// LDS layout is pre-swizzled into the exact WMMA fragment order, so each
// lane fetches its whole fragment with two ds_load_b128.
//   A frag (16-bit 16x32): k = (j&7) + 16*(j>>3) + 8*hu  -> group 2*kb+hu
//   B frag (16-bit 32x16): k = (j&7) + 16*(j>>3) + 8*hu  -> group 2*kb+hu
// =====================================================================
template<int ACT, bool HASBIAS, bool ACCUM>
__global__ __launch_bounds__(256) void gemm_wmma(
    const float* __restrict__ A, const float* __restrict__ W,
    const float* __restrict__ bias, float* __restrict__ C,
    int M, int N, int K) {
  __shared__ __align__(16) _Float16 As2[32][4][16];   // [row][group][frag]
  __shared__ __align__(16) _Float16 Bs2[64][4][16];   // [col][group][frag]
  const int m0 = blockIdx.x * 32, n0 = blockIdx.y * 64;
  const int tid = threadIdx.x, wave = tid >> 5, lane = tid & 31;
  const int wm = wave >> 2, wn = wave & 3;
  const int hu = lane >> 4;                    // lanes 16..31: second K half
  const int mrow = wm * 16 + (lane & 15);
  const int ncol = wn * 16 + (lane & 15);
  v8f acc = {};
  for (int kk = 0; kk < K; kk += 64) {
    // ---- load phase: all global loads issued first (b128) ----
    float4 ar[2], br[4];
#pragma unroll
    for (int it = 0; it < 2; ++it) {           // A tile 32x64 = 512 float4
      int idx = tid + it * 256;
      int r = idx >> 4, c4 = idx & 15;
      ar[it] = *(const float4*)&A[(size_t)(m0 + r) * K + kk + c4 * 4];
    }
#pragma unroll
    for (int it = 0; it < 4; ++it) {           // B tile 64x64 = 1024 float4
      int idx = tid + it * 256;
      int r = idx >> 4, c4 = idx & 15;
      br[it] = *(const float4*)&W[(size_t)(kk + r) * N + n0 + c4 * 4];
    }
    if (kk + 64 < K) {                         // CDNA5 global_prefetch_b8
      __builtin_prefetch(&A[(size_t)(m0 + (tid >> 3)) * K + kk + 64], 0, 1);
      __builtin_prefetch(&W[(size_t)(kk + 64 + (tid >> 2)) * N + n0], 0, 1);
    }
    // ---- store phase: convert + swizzle into fragment order ----
#pragma unroll
    for (int it = 0; it < 2; ++it) {
      int idx = tid + it * 256;
      int r = idx >> 4, c = (idx & 15) * 4;
      int cc = c & 31;
      int g = (c >> 5) * 2 + ((cc >> 3) & 1);
      int j0 = (cc & 7) + ((cc & 16) ? 8 : 0); // 4-aligned within frag
      v4h t = { (_Float16)ar[it].x, (_Float16)ar[it].y,
                (_Float16)ar[it].z, (_Float16)ar[it].w };
      *(v4h*)&As2[r][g][j0] = t;               // ds_store_b64
    }
#pragma unroll
    for (int it = 0; it < 4; ++it) {
      int idx = tid + it * 256;
      int r = idx >> 4, c = (idx & 15) * 4;
      int g = (r >> 5) * 2 + ((r >> 3) & 1);
      int j = (r & 7) + ((r & 16) ? 8 : 0);
      Bs2[c + 0][g][j] = (_Float16)br[it].x;
      Bs2[c + 1][g][j] = (_Float16)br[it].y;
      Bs2[c + 2][g][j] = (_Float16)br[it].z;
      Bs2[c + 3][g][j] = (_Float16)br[it].w;
    }
    __syncthreads();
    // ---- compute: 2 x (2 ds_load_b128 per operand + wmma) ----
#pragma unroll
    for (int kb = 0; kb < 2; ++kb) {
      int g = kb * 2 + hu;
      v8h a0 = *(const v8h*)&As2[mrow][g][0];
      v8h a1 = *(const v8h*)&As2[mrow][g][8];
      v8h b0 = *(const v8h*)&Bs2[ncol][g][0];
      v8h b1 = *(const v8h*)&Bs2[ncol][g][8];
      v16h av = __builtin_shufflevector(a0, a1, 0,1,2,3,4,5,6,7,8,9,10,11,12,13,14,15);
      v16h bv = __builtin_shufflevector(b0, b1, 0,1,2,3,4,5,6,7,8,9,10,11,12,13,14,15);
      acc = __builtin_amdgcn_wmma_f32_16x16x32_f16(false, av, false, bv,
                                                   (short)0, acc, false, false);
    }
    __syncthreads();
  }
  const int colg = n0 + ncol;
  float bv = 0.0f;
  if (HASBIAS) bv = bias[colg];
#pragma unroll
  for (int r = 0; r < 8; ++r) {                // D layout: M = r + 8*hu
    int rowg = m0 + wm * 16 + r + (hu << 3);
    float v = acc[r] + bv;
    if (ACT == 1) v = 0.5f * v * (1.0f + erff(v * 0.70710678118f));  // exact gelu
    size_t o = (size_t)rowg * N + colg;
    if (ACCUM) C[o] += v; else C[o] = v;
  }
}

// ---------------- LayerNorm: one wave32 per 512-wide row --------------
__global__ __launch_bounds__(256) void ln_kernel(
    const float* __restrict__ src, const float* __restrict__ gam,
    const float* __restrict__ bet, const float* __restrict__ pos,
    float* __restrict__ dst, float* __restrict__ dst2) {
  int row = blockIdx.x * 8 + (threadIdx.x >> 5);
  int lane = threadIdx.x & 31;
  const float* x = src + (size_t)row * cDIM;
  const float* pp = pos ? pos + (size_t)(row & (cT - 1)) * cDIM : nullptr;
  float v[16]; float s = 0.f;
#pragma unroll
  for (int i = 0; i < 16; ++i) {
    float t = x[lane + i * 32];
    if (pos) t += pp[lane + i * 32];
    v[i] = t; s += t;
  }
#pragma unroll
  for (int o = 16; o > 0; o >>= 1) s += __shfl_xor(s, o, 32);
  float mean = s * (1.0f / cDIM);
  float q = 0.f;
#pragma unroll
  for (int i = 0; i < 16; ++i) { float d = v[i] - mean; q += d * d; }
#pragma unroll
  for (int o = 16; o > 0; o >>= 1) q += __shfl_xor(q, o, 32);
  float rstd = rsqrtf(q * (1.0f / cDIM) + 1e-5f);
#pragma unroll
  for (int i = 0; i < 16; ++i) {
    int c = lane + i * 32;
    float y = (v[i] - mean) * rstd * gam[c] + bet[c];
    dst[(size_t)row * cDIM + c] = y;
    if (dst2) dst2[(size_t)row * cDIM + c] = y;
  }
}

// ---------------- deterministic gaussian rotations --------------------
__device__ inline unsigned pcg_hash(unsigned v) {
  v = v * 747796405u + 2891336453u;
  unsigned w = ((v >> ((v >> 28u) + 4u)) ^ v) * 277803737u;
  return (w >> 22u) ^ w;
}
__global__ void rotgen(float* __restrict__ rot, int seed) {
  int i = blockIdx.x * blockDim.x + threadIdx.x;
  if (i >= 4096) return;
  unsigned h1 = pcg_hash((unsigned)(i * 2 + 1) ^ (0x9e3779b9u * (unsigned)(seed + 1)));
  unsigned h2 = pcg_hash((unsigned)(i * 2 + 2) ^ (0x85ebca6bu * (unsigned)(seed + 1)));
  float u1 = ((float)h1 + 0.5f) * (1.0f / 4294967296.0f);
  float u2 = ((float)h2 + 0.5f) * (1.0f / 4294967296.0f);
  rot[i] = sqrtf(-2.0f * __logf(u1)) * __cosf(6.28318530718f * u2);
}

// ---------------- LSH bucketing: packed sort keys ---------------------
__global__ __launch_bounds__(64) void lsh_bucket(
    const float* __restrict__ QK, const float* __restrict__ rot,
    unsigned* __restrict__ skeys) {
  __shared__ __align__(16) float srot[4096];   // [d=64][h=4][r=16]
  int bh = blockIdx.x;
  int t = blockIdx.y * 64 + threadIdx.x;
  for (int i = threadIdx.x; i < 1024; i += 64)
    ((float4*)srot)[i] = ((const float4*)rot)[i];
  __syncthreads();
  int b = bh / cSH, s = bh % cSH;
  const float4* q4 = (const float4*)(QK + (size_t)(b * cT + t) * cDIM + (cLH + s) * cDH);
  float qv[64];
#pragma unroll
  for (int dd = 0; dd < 16; ++dd) {
    float4 z = q4[dd];
    qv[4 * dd + 0] = z.x; qv[4 * dd + 1] = z.y;
    qv[4 * dd + 2] = z.z; qv[4 * dd + 3] = z.w;
  }
  for (int h = 0; h < cNH; ++h) {
    float dots[16];
#pragma unroll
    for (int r = 0; r < 16; ++r) {
      float a = 0.f;
#pragma unroll
      for (int d = 0; d < 64; ++d) a += qv[d] * srot[d * 64 + h * 16 + r];
      dots[r] = a;
    }
    float best = -1e30f; int bi = 0;
#pragma unroll
    for (int idx = 0; idx < 32; ++idx) {       // argmax over [rot, -rot]
      float vv = (idx < 16) ? dots[idx] : -dots[idx - 16];
      if (vv > best) { best = vv; bi = idx; }
    }
    int j = h * cT + t;
    unsigned bandt = (unsigned)((bi + h * cNB) * cT + t);
    skeys[(size_t)bh * cNJ + j] = (bandt << 13) | (unsigned)j;   // total order => stable
  }
}

// ---------------- per-row bitonic sort (8192 u32 in LDS) --------------
__global__ __launch_bounds__(1024) void lsh_sort(
    const unsigned* __restrict__ skeys, int* __restrict__ sticker) {
  __shared__ unsigned sk[cNJ];
  int bh = blockIdx.x;
  for (int i = threadIdx.x; i < cNJ; i += 1024) sk[i] = skeys[(size_t)bh * cNJ + i];
  for (int k = 2; k <= cNJ; k <<= 1)
    for (int j = k >> 1; j > 0; j >>= 1) {
      __syncthreads();
      for (int i = threadIdx.x; i < cNJ; i += 1024) {
        int p = i ^ j;
        if (p > i) {
          unsigned a = sk[i], bb = sk[p];
          bool up = ((i & k) == 0);
          if (up ? (a > bb) : (a < bb)) { sk[i] = bb; sk[p] = a; }
        }
      }
    }
  __syncthreads();
  for (int i = threadIdx.x; i < cNJ; i += 1024)
    sticker[(size_t)bh * cNJ + i] = (int)(sk[i] & 0x1FFFu);      // original flat index
}

// ---------------- chunked LSH attention (look-back 1 chunk) -----------
__global__ __launch_bounds__(64) void lsh_chunk(
    const float* __restrict__ QK, const float* __restrict__ V,
    const int* __restrict__ sticker, const unsigned char* __restrict__ mask,
    float* __restrict__ O, float* __restrict__ LSE) {
  __shared__ __align__(16) _Float16 kx[128][68];
  __shared__ __align__(16) _Float16 vx[128][68];
  __shared__ __align__(16) float qx[64][68];
  __shared__ int ktk[128];
  __shared__ int kjf[64];
  __shared__ unsigned char kmk[128];
  int bh = blockIdx.x, c = blockIdx.y;
  int b = bh / cSH, s = bh % cSH;
  int head = cLH + s;
  for (int r = threadIdx.x; r < 128; r += 64) {
    int cc = (r < 64) ? c : ((c + cNCH - 1) & (cNCH - 1));       // roll(1) look-back
    int spos = cc * cBUCKET + (r & 63);
    int jf = sticker[(size_t)bh * cNJ + spos];
    int t = jf & (cT - 1);
    ktk[r] = t;
    kmk[r] = mask[b * cT + t];
    const float4* q4 = (const float4*)(QK + (size_t)(b * cT + t) * cDIM + head * cDH);
    const float4* v4p = (const float4*)(V + (size_t)(b * cT + t) * cDIM + head * cDH);
    float tmp[64]; float nn = 0.f;
#pragma unroll
    for (int dd = 0; dd < 16; ++dd) {
      float4 z = q4[dd];
      tmp[4 * dd + 0] = z.x; tmp[4 * dd + 1] = z.y;
      tmp[4 * dd + 2] = z.z; tmp[4 * dd + 3] = z.w;
      nn += z.x * z.x + z.y * z.y + z.z * z.z + z.w * z.w;
    }
    float inv = 1.0f / fmaxf(sqrtf(nn), 1e-12f);
#pragma unroll
    for (int dd = 0; dd < 16; ++dd) {
      v4h tk = { (_Float16)(tmp[4 * dd + 0] * inv), (_Float16)(tmp[4 * dd + 1] * inv),
                 (_Float16)(tmp[4 * dd + 2] * inv), (_Float16)(tmp[4 * dd + 3] * inv) };
      *(v4h*)&kx[r][4 * dd] = tk;              // l2-normalized keys (ds_store_b64)
      float4 w = v4p[dd];
      v4h tv = { (_Float16)w.x, (_Float16)w.y, (_Float16)w.z, (_Float16)w.w };
      *(v4h*)&vx[r][4 * dd] = tv;
    }
    if (r < 64) {
      kjf[r] = jf;
#pragma unroll
      for (int dd = 0; dd < 16; ++dd)          // raw queries, float4-aligned rows
        *(float4*)&qx[r][4 * dd] = make_float4(tmp[4*dd], tmp[4*dd+1], tmp[4*dd+2], tmp[4*dd+3]);
    }
  }
  __syncthreads();
  int i = threadIdx.x;
  float q[64];
#pragma unroll
  for (int dd = 0; dd < 16; ++dd) {
    float4 z = *(const float4*)&qx[i][4 * dd];
    q[4 * dd + 0] = z.x; q[4 * dd + 1] = z.y;
    q[4 * dd + 2] = z.z; q[4 * dd + 3] = z.w;
  }
  int ti = ktk[i];
  bool mi = kmk[i] != 0;
  float m = -INFINITY, l = 0.f, acc[64];
#pragma unroll
  for (int d = 0; d < 64; ++d) acc[d] = 0.f;
  for (int jx = 0; jx < 128; ++jx) {
    float dot = 0.f;
#pragma unroll
    for (int dd = 0; dd < 16; ++dd) {
      v4h k4 = *(const v4h*)&kx[jx][4 * dd];
      dot += q[4*dd+0] * (float)k4[0] + q[4*dd+1] * (float)k4[1]
           + q[4*dd+2] * (float)k4[2] + q[4*dd+3] * (float)k4[3];
    }
    dot *= 0.125f;                             // d^-0.5
    if (!(mi && kmk[jx])) dot = MASKV;         // mask first...
    if (ti == ktk[jx]) dot = SELFV;            // ...self overrides (reference order)
    float mn = fmaxf(m, dot);
    float corr = __expf(m - mn);
    float p = __expf(dot - mn);
    l = l * corr + p;
#pragma unroll
    for (int dd = 0; dd < 16; ++dd) {
      v4h w4 = *(const v4h*)&vx[jx][4 * dd];
      acc[4*dd+0] = acc[4*dd+0] * corr + p * (float)w4[0];
      acc[4*dd+1] = acc[4*dd+1] * corr + p * (float)w4[1];
      acc[4*dd+2] = acc[4*dd+2] * corr + p * (float)w4[2];
      acc[4*dd+3] = acc[4*dd+3] * corr + p * (float)w4[3];
    }
    m = mn;
  }
  float lse = m + __logf(l);
  float rl = 1.0f / l;
  int jf = kjf[i];
  int h = jf >> 11;                            // /T
  int t = jf & (cT - 1);
  float* op = O + (((size_t)bh * cNH + h) * cT + t) * cDH;       // scatter via sticker
#pragma unroll
  for (int d = 0; d < 64; ++d) op[d] = acc[d] * rl;
  LSE[((size_t)bh * cNH + h) * cT + t] = lse;
}

// ---------------- combine hash rounds by logsumexp softmax ------------
__global__ __launch_bounds__(64) void lsh_combine(
    const float* __restrict__ O, const float* __restrict__ LSE,
    float* __restrict__ att) {
  int bh = blockIdx.x, t = blockIdx.y, d = threadIdx.x;
  float lg[4];
#pragma unroll
  for (int h = 0; h < 4; ++h) lg[h] = LSE[((size_t)bh * cNH + h) * cT + t];
  float mx = fmaxf(fmaxf(lg[0], lg[1]), fmaxf(lg[2], lg[3]));
  float w[4]; float sum = 0.f;
#pragma unroll
  for (int h = 0; h < 4; ++h) { w[h] = __expf(lg[h] - mx); sum += w[h]; }
  float inv = 1.0f / sum;
  float o = 0.f;
#pragma unroll
  for (int h = 0; h < 4; ++h)
    o += w[h] * inv * O[(((size_t)bh * cNH + h) * cT + t) * cDH + d];
  int b = bh / cSH, s = bh % cSH;
  att[(size_t)(b * cT + t) * cDIM + (cLH + s) * cDH + d] = o;
}

// ---------------- local windowed attention (flash-style) --------------
__global__ __launch_bounds__(128) void local_attn(
    const float* __restrict__ QK, const float* __restrict__ V,
    const unsigned char* __restrict__ mask, float* __restrict__ att) {
  __shared__ __align__(16) _Float16 kx[128][68];
  __shared__ __align__(16) _Float16 vx[128][68];
  __shared__ int ktk[128];
  __shared__ unsigned char kmk[128];
  int bh = blockIdx.x, wdx = blockIdx.y;
  int b = bh / cLH, hl = bh % cLH;
  int i = threadIdx.x;
  int tq = wdx * cWIN + i;
  const float4* qr4 = (const float4*)(QK + (size_t)(b * cT + tq) * cDIM + hl * cDH);
  float q[64];
#pragma unroll
  for (int dd = 0; dd < 16; ++dd) {
    float4 z = qr4[dd];
    q[4 * dd + 0] = z.x; q[4 * dd + 1] = z.y;
    q[4 * dd + 2] = z.z; q[4 * dd + 3] = z.w;
  }
  bool mi = mask[b * cT + tq] != 0;
  float m = -INFINITY, l = 0.f, acc[64];
#pragma unroll
  for (int d = 0; d < 64; ++d) acc[d] = 0.f;
  for (int wo = -1; wo <= 1; ++wo) {           // prev, cur, next windows
    int wc = wdx + wo;
    __syncthreads();
    {
      int r = i;
      if (wc >= 0 && wc < cNW) {
        int tk = wc * cWIN + r;
        ktk[r] = tk;
        kmk[r] = mask[b * cT + tk];
        const float4* k4 = (const float4*)(QK + (size_t)(b * cT + tk) * cDIM + hl * cDH);
        const float4* w4 = (const float4*)(V + (size_t)(b * cT + tk) * cDIM + hl * cDH);
        float tmp[64]; float nn = 0.f;
#pragma unroll
        for (int dd = 0; dd < 16; ++dd) {
          float4 z = k4[dd];
          tmp[4 * dd + 0] = z.x; tmp[4 * dd + 1] = z.y;
          tmp[4 * dd + 2] = z.z; tmp[4 * dd + 3] = z.w;
          nn += z.x * z.x + z.y * z.y + z.z * z.z + z.w * z.w;
        }
        float inv = 1.0f / fmaxf(sqrtf(nn), 1e-12f);
#pragma unroll
        for (int dd = 0; dd < 16; ++dd) {
          v4h tk2 = { (_Float16)(tmp[4*dd]*inv), (_Float16)(tmp[4*dd+1]*inv),
                      (_Float16)(tmp[4*dd+2]*inv), (_Float16)(tmp[4*dd+3]*inv) };
          *(v4h*)&kx[r][4 * dd] = tk2;
          float4 z = w4[dd];
          v4h tv = { (_Float16)z.x, (_Float16)z.y, (_Float16)z.z, (_Float16)z.w };
          *(v4h*)&vx[r][4 * dd] = tv;
        }
      } else {
        ktk[r] = -1; kmk[r] = 0;
        v4h zz = { (_Float16)0.f, (_Float16)0.f, (_Float16)0.f, (_Float16)0.f };
#pragma unroll
        for (int dd = 0; dd < 16; ++dd) {
          *(v4h*)&kx[r][4 * dd] = zz; *(v4h*)&vx[r][4 * dd] = zz;
        }
      }
    }
    __syncthreads();
    for (int jx = 0; jx < 128; ++jx) {
      float dot = 0.f;
#pragma unroll
      for (int dd = 0; dd < 16; ++dd) {
        v4h k4 = *(const v4h*)&kx[jx][4 * dd];
        dot += q[4*dd+0] * (float)k4[0] + q[4*dd+1] * (float)k4[1]
             + q[4*dd+2] * (float)k4[2] + q[4*dd+3] * (float)k4[3];
      }
      dot *= 0.125f;
      if (tq == ktk[jx]) dot = SELFV;          // reference order: self, pad, mask
      if (ktk[jx] < 0) dot = MASKV;
      if (!(mi && kmk[jx])) dot = MASKV;
      float mn = fmaxf(m, dot);
      float corr = __expf(m - mn);
      float p = __expf(dot - mn);
      l = l * corr + p;
#pragma unroll
      for (int dd = 0; dd < 16; ++dd) {
        v4h w4 = *(const v4h*)&vx[jx][4 * dd];
        acc[4*dd+0] = acc[4*dd+0] * corr + p * (float)w4[0];
        acc[4*dd+1] = acc[4*dd+1] * corr + p * (float)w4[1];
        acc[4*dd+2] = acc[4*dd+2] * corr + p * (float)w4[2];
        acc[4*dd+3] = acc[4*dd+3] * corr + p * (float)w4[3];
      }
      m = mn;
    }
  }
  float rl = 1.0f / l;
  float* op = att + (size_t)(b * cT + tq) * cDIM + hl * cDH;
#pragma unroll
  for (int d = 0; d < 64; ++d) op[d] = acc[d] * rl;
}

// ---------------- product-key memory (x2 += pkm(gu)) ------------------
__global__ __launch_bounds__(128) void pkm_kernel(
    const float* __restrict__ Q, const float* __restrict__ keys,
    const float* __restrict__ values, float* __restrict__ x2) {
  __shared__ float outacc[cDIM];
  __shared__ float d1[128], d2[128];
  __shared__ float s1[32], s2[32];
  __shared__ int   i1[32], i2[32];
  __shared__ float cs[1024];
  __shared__ float fs[32];
  __shared__ int   fi[32];
  int tok = blockIdx.x, tid = threadIdx.x;
  for (int d = tid; d < cDIM; d += 128) outacc[d] = 0.f;
  for (int h = 0; h < 4; ++h) {
    __syncthreads();
    {
      const float4* qh = (const float4*)(Q + (size_t)tok * cDIM + h * 128);
      const float4* k1 = (const float4*)(keys + ((size_t)(h * 128 + tid) * 2 + 0) * 64);
      const float4* k2 = (const float4*)(keys + ((size_t)(h * 128 + tid) * 2 + 1) * 64);
      float a = 0.f, bq = 0.f;
#pragma unroll
      for (int dd = 0; dd < 16; ++dd) {
        float4 qa = qh[dd], qb = qh[16 + dd], ka = k1[dd], kb = k2[dd];
        a  += qa.x * ka.x + qa.y * ka.y + qa.z * ka.z + qa.w * ka.w;
        bq += qb.x * kb.x + qb.y * kb.y + qb.z * kb.z + qb.w * kb.w;
      }
      d1[tid] = a; d2[tid] = bq;
    }
    __syncthreads();
    {   // top-32 of 128 by rank (index-stable ties == jax top_k)
      float v = d1[tid]; int rank = 0;
      for (int u = 0; u < 128; ++u) { float w = d1[u]; rank += (w > v) || (w == v && u < tid); }
      if (rank < 32) { s1[rank] = v; i1[rank] = tid; }
      v = d2[tid]; rank = 0;
      for (int u = 0; u < 128; ++u) { float w = d2[u]; rank += (w > v) || (w == v && u < tid); }
      if (rank < 32) { s2[rank] = v; i2[rank] = tid; }
    }
    __syncthreads();
    for (int p = tid; p < 1024; p += 128) cs[p] = s1[p >> 5] + s2[p & 31];
    __syncthreads();
    for (int p = tid; p < 1024; p += 128) {
      float v = cs[p]; int rank = 0;
      for (int u = 0; u < 1024; ++u) { float w = cs[u]; rank += (w > v) || (w == v && u < p); }
      if (rank < 32) { fs[rank] = v; fi[rank] = i1[p >> 5] * 128 + i2[p & 31]; }
    }
    __syncthreads();
    float mx = fs[0], sum = 0.f, w[32];        // fs[0] == max (rank 0)
#pragma unroll
    for (int j = 0; j < 32; ++j) { w[j] = __expf(fs[j] - mx); sum += w[j]; }
    float inv = 1.0f / sum;
    for (int d = tid; d < cDIM; d += 128) {
      float a = 0.f;
#pragma unroll
      for (int j = 0; j < 32; ++j) a += w[j] * values[(size_t)fi[j] * cDIM + d];
      outacc[d] += a * inv;
    }
  }
  __syncthreads();
  for (int d = tid; d < cDIM; d += 128) x2[(size_t)tok * cDIM + d] += outacc[d];
}

// ---------------- epilogue: out = 0.5*(x1+x2) -------------------------
__global__ void final_avg(const float* __restrict__ a, const float* __restrict__ b2,
                          float* __restrict__ o, int n) {
  int i = blockIdx.x * 256 + threadIdx.x;
  if (i < n) o[i] = 0.5f * (a[i] + b2[i]);
}

// =====================================================================
extern "C" void kernel_launch(void* const* d_in, const int* in_sizes, int n_in,
                              void* d_out, int out_size, void* d_ws, size_t ws_size,
                              hipStream_t stream) {
  (void)in_sizes; (void)n_in; (void)out_size; (void)ws_size;
  const float* WH   = (const float*)d_in[0];
  const unsigned char* MASK = (const unsigned char*)d_in[1];
  const float* LINW = (const float*)d_in[2];
  const float* LINB = (const float*)d_in[3];
  const float* POS  = (const float*)d_in[4];
  const float* NG   = (const float*)d_in[5];
  const float* NBv  = (const float*)d_in[6];

  float* ws  = (float*)d_ws;
  float* X1  = ws + OFF_X1;
  float* X2  = ws + OFF_X2;
  float* H   = ws + OFF_H;
  float* QKb = ws + OFF_QK;
  float* Vb  = ws + OFF_V;
  float* ATT = ws + OFF_ATT;
  float* BIG = ws + OFF_BIG;
  float* Ob  = ws + OFF_O;
  float* LSEb= ws + OFF_LSE;
  float* ROT = ws + OFF_ROT;
  unsigned* SK = (unsigned*)(ws + OFF_SK);
  int* STk     = (int*)(ws + OFF_ST);

  dim3 blk(256);
  // input projection + pos + layernorm -> x1 = x2
  gemm_wmma<0, true, false><<<dim3(cROWS / 32, cDIM / 64), blk, 0, stream>>>(
      WH, LINW, LINB, H, cROWS, cDIM, cIN);
  ln_kernel<<<cROWS / 8, blk, 0, stream>>>(H, NG, NBv, POS, X1, X2);

  for (int L = 0; L < 2; ++L) {
    int pa = (L == 0) ? 7 : 19;
    const float* ALNG = (const float*)d_in[pa + 0];
    const float* ALNB = (const float*)d_in[pa + 1];
    const float* WQKm = (const float*)d_in[pa + 2];
    const float* WVm  = (const float*)d_in[pa + 3];
    const float* WOm  = (const float*)d_in[pa + 4];
    const float* BOv  = (const float*)d_in[pa + 5];

    // h = LN(x2); qk = h@wqk; v = h@wv
    ln_kernel<<<cROWS / 8, blk, 0, stream>>>(X2, ALNG, ALNB, nullptr, H, nullptr);
    gemm_wmma<0, false, false><<<dim3(cROWS / 32, cDIM / 64), blk, 0, stream>>>(
        H, WQKm, nullptr, QKb, cROWS, cDIM, cDIM);
    gemm_wmma<0, false, false><<<dim3(cROWS / 32, cDIM / 64), blk, 0, stream>>>(
        H, WVm, nullptr, Vb, cROWS, cDIM, cDIM);

    // local heads 0..1
    local_attn<<<dim3(cB * cLH, cNW), dim3(128), 0, stream>>>(QKb, Vb, MASK, ATT);

    // LSH heads 2..7
    rotgen<<<dim3(16), dim3(256), 0, stream>>>(ROT, L);
    lsh_bucket<<<dim3(cBhL, cT / 64), dim3(64), 0, stream>>>(QKb, ROT, SK);
    lsh_sort<<<dim3(cBhL), dim3(1024), 0, stream>>>(SK, STk);
    lsh_chunk<<<dim3(cBhL, cNCH), dim3(64), 0, stream>>>(QKb, Vb, STk, MASK, Ob, LSEb);
    lsh_combine<<<dim3(cBhL, cT), dim3(64), 0, stream>>>(Ob, LSEb, ATT);

    // x1 += attn @ wo + bo
    gemm_wmma<0, true, true><<<dim3(cROWS / 32, cDIM / 64), blk, 0, stream>>>(
        ATT, WOm, BOv, X1, cROWS, cDIM, cDIM);

    if (L == 0) {
      const float* GLNG = (const float*)d_in[13];
      const float* GLNB = (const float*)d_in[14];
      const float* W1   = (const float*)d_in[15];
      const float* B1   = (const float*)d_in[16];
      const float* W2   = (const float*)d_in[17];
      const float* B2   = (const float*)d_in[18];
      ln_kernel<<<cROWS / 8, blk, 0, stream>>>(X1, GLNG, GLNB, nullptr, H, nullptr);
      gemm_wmma<1, true, false><<<dim3(cROWS / 32, (4 * cDIM) / 64), blk, 0, stream>>>(
          H, W1, B1, BIG, cROWS, 4 * cDIM, cDIM);                 // gelu fused
      gemm_wmma<0, true, true><<<dim3(cROWS / 32, cDIM / 64), blk, 0, stream>>>(
          BIG, W2, B2, X2, cROWS, cDIM, 4 * cDIM);                // x2 +=
    } else {
      const float* GLNG = (const float*)d_in[25];
      const float* GLNB = (const float*)d_in[26];
      const float* WQm  = (const float*)d_in[27];
      const float* KEYS = (const float*)d_in[28];
      const float* VALS = (const float*)d_in[29];
      ln_kernel<<<cROWS / 8, blk, 0, stream>>>(X1, GLNG, GLNB, nullptr, H, nullptr);
      gemm_wmma<0, false, false><<<dim3(cROWS / 32, cDIM / 64), blk, 0, stream>>>(
          H, WQm, nullptr, QKb, cROWS, cDIM, cDIM);
      pkm_kernel<<<dim3(cROWS), dim3(128), 0, stream>>>(QKb, KEYS, VALS, X2);
    }
  }
  final_avg<<<dim3((cROWS * cDIM + 255) / 256), dim3(256), 0, stream>>>(
      X1, X2, (float*)d_out, cROWS * cDIM);
}
